// VGRNN_7851200217454
// MI455X (gfx1250) — matrix operand order repaired and checked by reading, not compiled, not run on hardware
//
#include <hip/hip_runtime.h>
#include <hip/hip_bf16.h>
#include <math.h>

#define T_STEPS 4
#define NN      4096
#define XDIM    256
#define HDIM    256
#define ZDIM    128
#define NE      131072
#define EPSF    1e-10f

typedef __attribute__((ext_vector_type(16))) _Float16 v16h;
typedef __attribute__((ext_vector_type(8)))  float    v8f;

__device__ __forceinline__ float softplusf(float x) {
    return fmaxf(x, 0.f) + log1pf(expf(-fabsf(x)));
}

__device__ __forceinline__ float wave_reduce(float v) {
    #pragma unroll
    for (int i = 16; i >= 1; i >>= 1) v += __shfl_xor(v, i, 32);
    return v;
}

__device__ __forceinline__ v8f wmma_f16(const v16h& a, const v16h& b, const v8f& c) {
    return __builtin_amdgcn_wmma_f32_16x16x32_f16(false, a, false, b,
                                                  (short)0, c, false, false);
}

// A fragment (contiguous K=ka..ka+15, ka = 16*(lane/16)) from f32 row.
__device__ __forceinline__ v16h load_a_frag_f32(const float* __restrict__ ap) {
    v16h a;
    #pragma unroll
    for (int j = 0; j < 8; ++j) {
        a[j]     = (_Float16)ap[j];
        a[j + 8] = (_Float16)ap[j + 8];
    }
    return a;
}

// Same fragment from an f16 row (two contiguous 16B chunks -> b128 loads).
__device__ __forceinline__ v16h load_a_frag_f16(const _Float16* __restrict__ ap) {
    v16h a;
    #pragma unroll
    for (int j = 0; j < 8; ++j) {
        a[j]     = ap[j];
        a[j + 8] = ap[j + 8];
    }
    return a;
}

// ---------------------------------------------------------------------------
// WMMA GEMM: C[M,N] = act( A[M,K] @ B[K,N] (+C if ACCUM) (+bias if HAS_BIAS) )
// A: row-major f32.  Bt: weights pre-transposed+converted to f16, layout N x K
// (ldbt = K span) so each lane's B fragment is one contiguous 32-byte load.
// One wave computes a 32x16 C tile (2 WMMAs / K-step, B fragment reused).
// M % 32 == 0, N % 16 == 0, K % 32 == 0.  ACT: 0=none, 1=relu, 2=softplus
// ---------------------------------------------------------------------------
template<int ACCUM, int HAS_BIAS, int ACT>
__global__ __launch_bounds__(256)
void gemm_f16wmma(const float* __restrict__ A, int lda,
                  const _Float16* __restrict__ Bt, int ldbt,
                  float* __restrict__ C, int ldc,
                  int M, int N, int K, const float* __restrict__ bias)
{
    const int lane = threadIdx.x & 31;
    const int wave = blockIdx.x * (blockDim.x >> 5) + (threadIdx.x >> 5);
    const int tilesN = N >> 4;
    const int totalTiles = (M >> 5) * tilesN;
    if (wave >= totalTiles) return;               // wave-uniform: EXEC stays all-1s
    const int tm = (wave / tilesN) << 5;
    const int tn = (wave % tilesN) << 4;

    const int m  = lane & 15;
    const int ka = (lane >> 4) << 4;              // A k-offset: 0 or 16
    const int kb = (lane >> 4) << 4;              // B k-offset: 0 or 16
    const int n  = lane & 15;

    const _Float16* bbase = Bt + (size_t)(tn + n) * ldbt + kb;
    v8f acc0 = {}, acc1 = {};
    for (int k0 = 0; k0 < K; k0 += 32) {
        v16h bv = *reinterpret_cast<const v16h*>(bbase + k0);   // 32B contiguous
        const float* ap0 = A + (size_t)(tm + m) * lda + k0 + ka;
        acc0 = wmma_f16(load_a_frag_f32(ap0), bv, acc0);
        acc1 = wmma_f16(load_a_frag_f32(ap0 + (size_t)16 * lda), bv, acc1);
    }

    const int row0 = tm + ((lane >> 4) << 3);     // lanes 16-31 hold M=+8
    const int col  = tn + n;
    const float bval = HAS_BIAS ? bias[col] : 0.f;
    #pragma unroll
    for (int r = 0; r < 8; ++r) {
        #pragma unroll
        for (int s = 0; s < 2; ++s) {
            float* cp = C + (size_t)(row0 + s * 16 + r) * ldc + col;
            float v = (s ? acc1[r] : acc0[r]) + bval;
            if (ACCUM) v += *cp;
            if (ACT == 1)      v = fmaxf(v, 0.f);
            else if (ACT == 2) v = softplusf(v);
            *cp = v;
        }
    }
}

// ---------------------------------------------------------------------------
// Fused inner-product decoder + weighted-BCE partial sums.
// logits = z @ z.T from the f16 shadow copy zh (NN x ZDIM): both fragments are
// packed-f16 contiguous loads, zero conversions in the inner loop. One wave
// computes a 64x16 logits tile (4 WMMAs / K-step, B fragment reused 4x), folds
// adj in-register, wave-reduces [sum_a, sum_a*sp(-l), sum_(1-a)*sp(l)].
// ---------------------------------------------------------------------------
__global__ __launch_bounds__(256)
void decoder_bce(const _Float16* __restrict__ zh,
                 const float* __restrict__ adj,
                 float* __restrict__ acc3)
{
    const int lane = threadIdx.x & 31;
    const int wave = blockIdx.x * (blockDim.x >> 5) + (threadIdx.x >> 5);
    const int tilesN = NN >> 4;                   // 256
    const int tm = (wave / tilesN) << 6;          // 64-row block
    const int tn = (wave % tilesN) << 4;

    // Prefetch the adj tile (dominant HBM stream) under the WMMA loop.
    __builtin_prefetch(adj + (size_t)(tm + lane) * NN + tn, 0, 3);
    __builtin_prefetch(adj + (size_t)(tm + 32 + lane) * NN + tn, 0, 3);

    const int m  = lane & 15;
    const int ka = (lane >> 4) << 4;
    const int kb = (lane >> 4) << 4;
    const int n  = lane & 15;

    const _Float16* bbase = zh + (size_t)(tn + n) * ZDIM + kb;  // B[k][n]=z[n][k]
    v8f acc[4] = {};
    #pragma unroll
    for (int k0 = 0; k0 < ZDIM; k0 += 32) {
        v16h bv = *reinterpret_cast<const v16h*>(bbase + k0);
        #pragma unroll
        for (int s = 0; s < 4; ++s) {
            const _Float16* ap = zh + (size_t)(tm + s * 16 + m) * ZDIM + k0 + ka;
            acc[s] = wmma_f16(load_a_frag_f16(ap), bv, acc[s]);
        }
    }

    const int rbase = ((lane >> 4) << 3);
    const int col   = tn + n;
    float sa = 0.f, s1 = 0.f, s2 = 0.f;
    #pragma unroll
    for (int s = 0; s < 4; ++s) {
        const int row0 = tm + s * 16 + rbase;
        #pragma unroll
        for (int r = 0; r < 8; ++r) {
            float l  = acc[s][r];
            float a  = adj[(size_t)(row0 + r) * NN + col];
            float sp = softplusf(l);              // softplus(-l) = sp - l
            sa += a;
            s1 += a * (sp - l);
            s2 += (1.f - a) * sp;
        }
    }
    sa = wave_reduce(sa); s1 = wave_reduce(s1); s2 = wave_reduce(s2);
    if (lane == 0) {
        atomicAdd(acc3 + 0, sa);
        atomicAdd(acc3 + 1, s1);
        atomicAdd(acc3 + 2, s2);
    }
}

// --------------------------- small helper kernels ---------------------------
__global__ void fill_kernel(float* p, float v, int n) {
    int i = blockIdx.x * blockDim.x + threadIdx.x;
    if (i < n) p[i] = v;
}

// W (K x N, f32, row-major) -> Wt (N x K, f16): one-time weight prep.
__global__ void transpose_to_f16(const float* __restrict__ W, _Float16* __restrict__ Wt,
                                 int K, int N) {
    int idx = blockIdx.x * blockDim.x + threadIdx.x;
    if (idx >= K * N) return;
    int k = idx / N, n = idx % N;                 // coalesced read over n
    Wt[(size_t)n * K + k] = (_Float16)W[idx];
}

__global__ void deg_count(const int* __restrict__ dst, float* __restrict__ deg) {
    int e = blockIdx.x * blockDim.x + threadIdx.x;
    if (e < NE) atomicAdd(&deg[dst[e]], 1.f);
}

__global__ void deg_finalize(const float* __restrict__ deg,
                             float* __restrict__ dinv, float* __restrict__ invdeg) {
    int i = blockIdx.x * blockDim.x + threadIdx.x;
    if (i < NN) {
        float d = deg[i] + 1.f;                   // +1 self loop
        dinv[i]   = rsqrtf(d);
        invdeg[i] = 1.f / d;
    }
}

// out = xw * invdeg[row] + bias   (self-loop term, pre-scatter init)
__global__ void gcn_init(const float* __restrict__ xw, const float* __restrict__ invdeg,
                         const float* __restrict__ bias, float* __restrict__ out, int C) {
    int idx = blockIdx.x * blockDim.x + threadIdx.x;
    if (idx >= NN * C) return;
    int i = idx / C, c = idx % C;
    out[idx] = xw[idx] * invdeg[i] + (bias ? bias[c] : 0.f);
}

// out[dst] += xw[src] * dinv[src]*dinv[dst], float4-vectorized atomics
__global__ void gcn_scatter(const float* __restrict__ xw, const int* __restrict__ src,
                            const int* __restrict__ dst, const float* __restrict__ dinv,
                            float* __restrict__ out, int C) {
    int idx = blockIdx.x * blockDim.x + threadIdx.x;
    int per = C >> 2;
    if (idx >= NE * per) return;
    int e = idx / per;
    int c = (idx % per) << 2;
    int s = src[e], d = dst[e];
    float w = dinv[s] * dinv[d];
    const float4 v = *reinterpret_cast<const float4*>(xw + (size_t)s * C + c);
    float* o = out + (size_t)d * C + c;
    atomicAdd(o + 0, v.x * w);
    atomicAdd(o + 1, v.y * w);
    atomicAdd(o + 2, v.z * w);
    atomicAdd(o + 3, v.w * w);
}

// act: 2=softplus, 3=sigmoid, 4=tanh (in place)
__global__ void ew_act(float* p, int n, int act) {
    int i = blockIdx.x * blockDim.x + threadIdx.x;
    if (i >= n) return;
    float v = p[i];
    if (act == 2)      v = softplusf(v);
    else if (act == 3) v = 1.f / (1.f + expf(-v));
    else if (act == 4) v = tanhf(v);
    p[i] = v;
}

// z = em + noise*es, plus f16 shadow copy for the WMMA decoder
__global__ void ew_z(const float* em, const float* es, const float* nz,
                     float* z, _Float16* zh, int n) {
    int i = blockIdx.x * blockDim.x + threadIdx.x;
    if (i < n) {
        float v = em[i] + nz[i] * es[i];
        z[i]  = v;
        zh[i] = (_Float16)v;
    }
}

__global__ void ew_mul(const float* a, const float* b, float* o, int n) {
    int i = blockIdx.x * blockDim.x + threadIdx.x;
    if (i < n) o[i] = a[i] * b[i];
}

__global__ void ew_gru(const float* zg, const float* htl, float* h, int n) {
    int i = blockIdx.x * blockDim.x + threadIdx.x;
    if (i < n) h[i] = zg[i] * h[i] + (1.f - zg[i]) * htl[i];
}

__global__ void kld_kernel(const float* __restrict__ em, const float* __restrict__ es,
                           const float* __restrict__ pm, const float* __restrict__ ps,
                           float* __restrict__ out0) {
    int i = blockIdx.x * blockDim.x + threadIdx.x;
    float v = 0.f;
    if (i < NN * ZDIM) {
        float e = es[i] + EPSF, p = ps[i] + EPSF;
        float dm = em[i] - pm[i];
        v = 2.f * (logf(p) - logf(e)) + (e * e + dm * dm) / (p * p) - 1.f;
    }
    v = wave_reduce(v);
    if ((threadIdx.x & 31) == 0)
        atomicAdd(out0, v * (0.5f / ((float)NN * (float)NN)));
}

__global__ void nll_finalize(const float* __restrict__ acc3, float* __restrict__ out1) {
    float s  = acc3[0], s1 = acc3[1], s2 = acc3[2];
    float n2 = (float)NN * (float)NN;
    float posw = (n2 - s) / s;
    float norm = n2 / (2.f * (n2 - s));
    *out1 += norm * (posw * s1 + s2) / n2;
}

// ------------------------- host-side template dispatch -----------------------
static void launch_gemm(const float* A, int lda, const _Float16* Bt, int ldbt,
                        float* C, int ldc, int M, int N, int K,
                        const float* bias, int accum, int act, hipStream_t s) {
    int tiles = (M >> 5) * (N >> 4);
    int blocks = (tiles + 7) / 8;                 // 8 waves / 256-thread block
#define GL(AC, BI, ACT) gemm_f16wmma<AC, BI, ACT><<<blocks, 256, 0, s>>>( \
        A, lda, Bt, ldbt, C, ldc, M, N, K, bias)
    if (accum)         GL(1, 0, 0);               // accum chains: no bias / no act
    else if (bias) {
        if (act == 1)      GL(0, 1, 1);
        else if (act == 2) GL(0, 1, 2);
        else               GL(0, 1, 0);
    } else {
        if (act == 1)      GL(0, 0, 1);
        else if (act == 2) GL(0, 0, 2);
        else               GL(0, 0, 0);
    }
#undef GL
}

// ---------------------------------------------------------------------------
extern "C" void kernel_launch(void* const* d_in, const int* in_sizes, int n_in,
                              void* d_out, int out_size, void* d_ws, size_t ws_size,
                              hipStream_t stream) {
    const float* x      = (const float*)d_in[0];
    const int*   ei     = (const int*)  d_in[1];
    const float* adj    = (const float*)d_in[2];
    const float* noise  = (const float*)d_in[3];
    const float* w_phi_x = (const float*)d_in[4];  const float* b_phi_x = (const float*)d_in[5];
    const float* w_phi_z = (const float*)d_in[6];  const float* b_phi_z = (const float*)d_in[7];
    const float* w_enc   = (const float*)d_in[8];  const float* b_enc   = (const float*)d_in[9];
    const float* w_enc_mean = (const float*)d_in[10]; const float* b_enc_mean = (const float*)d_in[11];
    const float* w_enc_std  = (const float*)d_in[12]; const float* b_enc_std  = (const float*)d_in[13];
    const float* w_prior      = (const float*)d_in[14]; const float* b_prior      = (const float*)d_in[15];
    const float* w_prior_mean = (const float*)d_in[16]; const float* b_prior_mean = (const float*)d_in[17];
    const float* w_prior_std  = (const float*)d_in[18]; const float* b_prior_std  = (const float*)d_in[19];
    const float* w_xz = (const float*)d_in[20]; const float* w_hz = (const float*)d_in[21];
    const float* w_xr = (const float*)d_in[22]; const float* w_hr = (const float*)d_in[23];
    const float* w_xh = (const float*)d_in[24]; const float* w_hh = (const float*)d_in[25];

    float* out = (float*)d_out;

    // ---- workspace layout ----
    float* ws = (float*)d_ws;
    size_t off = 0;
    auto alloc = [&](size_t n) { float* p = ws + off; off += n; return p; };
    auto alloc_h = [&](size_t n) { return (_Float16*)alloc((n + 1) / 2); };
    const size_t NH = (size_t)NN * HDIM, NZ = (size_t)NN * ZDIM;
    float* h        = alloc(NH);
    float* phi_x    = alloc(NH);
    float* phi_z    = alloc(NH);
    float* enc_h    = alloc(NH);
    float* prior_h  = alloc(NH);
    float* tmpA     = alloc(NH);   // GEMM accumulation target (xw)
    float* zg       = alloc(NH);
    float* rg       = alloc(NH);
    float* rh       = alloc(NH);
    float* htl      = alloc(NH);
    float* enc_mean = alloc(NZ);
    float* enc_std  = alloc(NZ);
    float* prior_mean = alloc(NZ);
    float* prior_std  = alloc(NZ);
    float* zbuf     = alloc(NZ);
    float* tmpZ     = alloc(NZ);   // xw for 128-wide gcn
    float* deg      = alloc(NN);
    float* dinv     = alloc(NN);
    float* invdeg   = alloc(NN);
    float* acc3     = alloc(4);
    _Float16* zh    = alloc_h(NZ);
    // pre-transposed f16 weights (N x K each)
    _Float16* wt_phi_x      = alloc_h(256 * 256);
    _Float16* wt_phi_z      = alloc_h(128 * 256);
    _Float16* wt_enc        = alloc_h(512 * 256);
    _Float16* wt_enc_mean   = alloc_h(256 * 128);
    _Float16* wt_enc_std    = alloc_h(256 * 128);
    _Float16* wt_prior      = alloc_h(256 * 256);
    _Float16* wt_prior_mean = alloc_h(256 * 128);
    _Float16* wt_prior_std  = alloc_h(256 * 128);
    _Float16* wt_xz         = alloc_h(512 * 256);
    _Float16* wt_hz         = alloc_h(256 * 256);
    _Float16* wt_xr         = alloc_h(512 * 256);
    _Float16* wt_hr         = alloc_h(256 * 256);
    _Float16* wt_xh         = alloc_h(512 * 256);
    _Float16* wt_hh         = alloc_h(256 * 256);

    auto gemm = [&](const float* A, int lda, const _Float16* Bt, int ldbt,
                    float* C, int ldc, int M, int N, int K,
                    const float* bias, int accum, int act) {
        launch_gemm(A, lda, Bt, ldbt, C, ldc, M, N, K, bias, accum, act, stream);
    };
    auto fill = [&](float* p, float v, int n) {
        fill_kernel<<<(n + 255) / 256, 256, 0, stream>>>(p, v, n);
    };
    auto prep_w = [&](const float* W, _Float16* Wt, int K, int N) {
        transpose_to_f16<<<(K * N + 255) / 256, 256, 0, stream>>>(W, Wt, K, N);
    };
    auto gcn_agg = [&](const float* xw, const float* bias, float* dst_buf, int C,
                       const int* srcI, const int* dstI) {
        int n = NN * C;
        gcn_init<<<(n + 255) / 256, 256, 0, stream>>>(xw, invdeg, bias, dst_buf, C);
        int tot = NE * (C >> 2);
        gcn_scatter<<<(tot + 255) / 256, 256, 0, stream>>>(xw, srcI, dstI, dinv, dst_buf, C);
    };

    // ---- one-time weight prep: transpose + convert to f16 ----
    prep_w(w_phi_x,      wt_phi_x,      256, 256);
    prep_w(w_phi_z,      wt_phi_z,      128, 256);
    prep_w(w_enc,        wt_enc,        512, 256);
    prep_w(w_enc_mean,   wt_enc_mean,   256, 128);
    prep_w(w_enc_std,    wt_enc_std,    256, 128);
    prep_w(w_prior,      wt_prior,      256, 256);
    prep_w(w_prior_mean, wt_prior_mean, 256, 128);
    prep_w(w_prior_std,  wt_prior_std,  256, 128);
    prep_w(w_xz,         wt_xz,         512, 256);
    prep_w(w_hz,         wt_hz,         256, 256);
    prep_w(w_xr,         wt_xr,         512, 256);
    prep_w(w_hr,         wt_hr,         256, 256);
    prep_w(w_xh,         wt_xh,         512, 256);
    prep_w(w_hh,         wt_hh,         256, 256);

    fill(out, 0.f, 2);
    fill(h, 0.f, (int)NH);

    for (int t = 0; t < T_STEPS; ++t) {
        const float* x_t     = x     + (size_t)t * NN * XDIM;
        const int*   srcI    = ei    + (size_t)t * 2 * NE;
        const int*   dstI    = srcI  + NE;
        const float* adj_t   = adj   + (size_t)t * NN * NN;
        const float* noise_t = noise + (size_t)t * NN * ZDIM;

        // degree / normalization for this timestep's graph
        fill(deg, 0.f, NN);
        deg_count<<<(NE + 255) / 256, 256, 0, stream>>>(dstI, deg);
        deg_finalize<<<(NN + 255) / 256, 256, 0, stream>>>(deg, dinv, invdeg);

        // phi_x = relu(x @ w_phi_x + b)
        gemm(x_t, XDIM, wt_phi_x, 256, phi_x, HDIM, NN, HDIM, XDIM, b_phi_x, 0, 1);

        // enc_h = gcn(concat(phi_x,h) @ w_enc) + b_enc   (K-slices of wt_enc)
        gemm(phi_x, HDIM, wt_enc,       512, tmpA, HDIM, NN, HDIM, HDIM, nullptr, 0, 0);
        gemm(h,     HDIM, wt_enc + 256, 512, tmpA, HDIM, NN, HDIM, HDIM, nullptr, 1, 0);
        gcn_agg(tmpA, b_enc, enc_h, HDIM, srcI, dstI);

        // enc_mean / enc_std
        gemm(enc_h, HDIM, wt_enc_mean, 256, tmpZ, ZDIM, NN, ZDIM, HDIM, nullptr, 0, 0);
        gcn_agg(tmpZ, b_enc_mean, enc_mean, ZDIM, srcI, dstI);
        gemm(enc_h, HDIM, wt_enc_std, 256, tmpZ, ZDIM, NN, ZDIM, HDIM, nullptr, 0, 0);
        gcn_agg(tmpZ, b_enc_std, enc_std, ZDIM, srcI, dstI);
        ew_act<<<((int)NZ + 255) / 256, 256, 0, stream>>>(enc_std, (int)NZ, 2);

        // prior
        gemm(h, HDIM, wt_prior, 256, prior_h, HDIM, NN, HDIM, HDIM, b_prior, 0, 1);
        gemm(prior_h, HDIM, wt_prior_mean, 256, prior_mean, ZDIM, NN, ZDIM, HDIM, b_prior_mean, 0, 0);
        gemm(prior_h, HDIM, wt_prior_std,  256, prior_std,  ZDIM, NN, ZDIM, HDIM, b_prior_std,  0, 2);

        // z = enc_mean + noise * enc_std (+ f16 shadow) ; KLD
        ew_z<<<((int)NZ + 255) / 256, 256, 0, stream>>>(enc_mean, enc_std, noise_t,
                                                        zbuf, zh, (int)NZ);
        kld_kernel<<<((int)NZ + 255) / 256, 256, 0, stream>>>(enc_mean, enc_std,
                                                              prior_mean, prior_std, out + 0);

        // phi_z = relu(z @ w_phi_z + b)
        gemm(zbuf, ZDIM, wt_phi_z, 128, phi_z, HDIM, NN, HDIM, ZDIM, b_phi_z, 0, 1);

        // fused decoder + BCE: 64x16 tiles -> (4096/64)*(4096/16) waves
        fill(acc3, 0.f, 4);
        {
            int waves = (NN >> 6) * (NN >> 4);
            decoder_bce<<<(waves + 7) / 8, 256, 0, stream>>>(zh, adj_t, acc3);
        }
        nll_finalize<<<1, 1, 0, stream>>>(acc3, out + 1);

        // GRU gates: aggregation is linear -> sum GEMMs, single scatter per gate
        gemm(phi_x, HDIM, wt_xz,       512, tmpA, HDIM, NN, HDIM, HDIM, nullptr, 0, 0);
        gemm(phi_z, HDIM, wt_xz + 256, 512, tmpA, HDIM, NN, HDIM, HDIM, nullptr, 1, 0);
        gemm(h,     HDIM, wt_hz,       256, tmpA, HDIM, NN, HDIM, HDIM, nullptr, 1, 0);
        gcn_agg(tmpA, nullptr, zg, HDIM, srcI, dstI);
        ew_act<<<((int)NH + 255) / 256, 256, 0, stream>>>(zg, (int)NH, 3);

        gemm(phi_x, HDIM, wt_xr,       512, tmpA, HDIM, NN, HDIM, HDIM, nullptr, 0, 0);
        gemm(phi_z, HDIM, wt_xr + 256, 512, tmpA, HDIM, NN, HDIM, HDIM, nullptr, 1, 0);
        gemm(h,     HDIM, wt_hr,       256, tmpA, HDIM, NN, HDIM, HDIM, nullptr, 1, 0);
        gcn_agg(tmpA, nullptr, rg, HDIM, srcI, dstI);
        ew_act<<<((int)NH + 255) / 256, 256, 0, stream>>>(rg, (int)NH, 3);

        ew_mul<<<((int)NH + 255) / 256, 256, 0, stream>>>(rg, h, rh, (int)NH);

        gemm(phi_x, HDIM, wt_xh,       512, tmpA, HDIM, NN, HDIM, HDIM, nullptr, 0, 0);
        gemm(phi_z, HDIM, wt_xh + 256, 512, tmpA, HDIM, NN, HDIM, HDIM, nullptr, 1, 0);
        gemm(rh,    HDIM, wt_hh,       256, tmpA, HDIM, NN, HDIM, HDIM, nullptr, 1, 0);
        gcn_agg(tmpA, nullptr, htl, HDIM, srcI, dstI);
        ew_act<<<((int)NH + 255) / 256, 256, 0, stream>>>(htl, (int)NH, 4);

        ew_gru<<<((int)NH + 255) / 256, 256, 0, stream>>>(zg, htl, h, (int)NH);
    }
    (void)in_sizes; (void)n_in; (void)out_size; (void)ws_size;
}
